// DeltaMLModel_85873576116382
// MI455X (gfx1250) — matrix operand order repaired and checked by reading, not compiled, not run on hardware
//
#include <hip/hip_runtime.h>

// Problem constants (from reference)
#define Nn 50000
#define Ee 400000
#define Bb 8
#define Hh 128
#define Ll 3

typedef unsigned int u32;
typedef unsigned short u16;
typedef __attribute__((ext_vector_type(16))) __bf16 v16bf;
typedef __attribute__((ext_vector_type(8)))  float  v8f;

struct alignas(16) Q { u32 x, y, z, w; };
struct QQ { Q a, b; };
struct alignas(16) F4 { float x, y, z, w; };

__device__ __forceinline__ u16 f2bf(float f) {
  u32 u = __float_as_uint(f);
  u32 r = u + 0x7FFFu + ((u >> 16) & 1u);   // round-to-nearest-even
  return (u16)(r >> 16);
}
__device__ __forceinline__ u32 pack2(float a, float b) {
  return (u32)f2bf(a) | ((u32)f2bf(b) << 16);
}
// A operand fragment: lane holds row (lane&15); two contiguous 8xbf16 chunks
// at element offsets [off, off+8) and [off+16, off+24) of the row.
__device__ __forceinline__ v16bf frag_from_rowmajor(const u16* p) {
  QQ t;
  t.a = *(const Q*)p;
  t.b = *(const Q*)(p + 16);
  return __builtin_bit_cast(v16bf, t);
}
// B operand fragment: 8 contiguous dwords per lane from the packed buffer.
__device__ __forceinline__ v16bf frag_from_packed(const u32* p) {
  QQ t;
  t.a = *(const Q*)p;
  t.b = *(const Q*)(p + 4);
  return __builtin_bit_cast(v16bf, t);
}
__device__ __forceinline__ v8f wmma_bf16(v16bf a, v16bf b, v8f c) {
  return __builtin_amdgcn_wmma_f32_16x16x32_bf16(false, a, false, b,
                                                 (short)0, c, false, false);
}
__device__ __forceinline__ v8f vzero8() {
  v8f z;
#pragma unroll
  for (int j = 0; j < 8; ++j) z[j] = 0.0f;
  return z;
}

// ---------------------------------------------------------------------------
// Pack a row-major f32 weight W[K,N] into WMMA bf16 B-operand fragments.
// Fragment (kk,nt): lane L holds column n = nt*16 + (L%16), K range
// kbase = kk*32 + (L<16 ? 0 : 8), dword j = bf16 pair at kbase + 2*(j&3) + 16*(j>>2).
// Output dword index = ((kk*NT + nt)*32 + L)*8 + j  (== gid).
// ---------------------------------------------------------------------------
__global__ void pack_wmma_b(const float* __restrict__ W, u32* __restrict__ out,
                            int K, int N) {
  int gid = blockIdx.x * 256 + threadIdx.x;
  int total = (K / 32) * (N / 16) * 256;
  if (gid >= total) return;
  int j    = gid & 7;
  int lane = (gid >> 3) & 31;
  int frag = gid >> 8;
  int NT   = N >> 4;
  int nt   = frag % NT;
  int kk   = frag / NT;
  int n  = nt * 16 + (lane & 15);
  int kb = kk * 32 + ((lane < 16) ? 0 : 8);
  int k  = kb + 2 * (j & 3) + ((j >> 2) << 4);
  out[gid] = pack2(W[(size_t)k * N + n], W[(size_t)(k + 1) * N + n]);
}

// ---------------------------------------------------------------------------
// Embeddings
// ---------------------------------------------------------------------------
__global__ void embed_nodes_kernel(const float* __restrict__ x,
                                   const float* __restrict__ Wn,
                                   const float* __restrict__ bn,
                                   float* __restrict__ h,
                                   u16* __restrict__ hbf) {
  int gid = blockIdx.x * 256 + threadIdx.x;
  if (gid >= Nn * Hh) return;
  int n = gid >> 7, c = gid & 127;
  float s = bn[c];
  const float* xr = x + (size_t)n * 32;
#pragma unroll
  for (int k = 0; k < 32; ++k) s += xr[k] * Wn[k * Hh + c];
  h[gid]   = s;
  hbf[gid] = f2bf(s);
}

__global__ void embed_edges_kernel(const float* __restrict__ ea,
                                   const float* __restrict__ We,
                                   const float* __restrict__ be,
                                   u16* __restrict__ ebf) {
  int gid = blockIdx.x * 256 + threadIdx.x;
  if (gid >= Ee * Hh) return;
  int e = gid >> 7, c = gid & 127;
  float s = be[c];
  const float* er = ea + (size_t)e * 16;
#pragma unroll
  for (int k = 0; k < 16; ++k) s += er[k] * We[k * Hh + c];
  ebf[gid] = f2bf(s);
}

__global__ void kfill0(float* __restrict__ p, int n) {
  int gid = blockIdx.x * 256 + threadIdx.x;
  if (gid < n) p[gid] = 0.0f;
}

// ---------------------------------------------------------------------------
// Fused edge MLP + scatter-add, M=32 per wave (two 16-row tiles share every
// B fragment; B double-buffered in registers):
//   m = relu([h[dst]|h[src]|e] @ W1 + b1) @ W2 + b2 ; aggr[dst] += m
// ---------------------------------------------------------------------------
__global__ __launch_bounds__(128) void edge_mlp_kernel(
    const u16* __restrict__ hbf, const u16* __restrict__ ebf,
    const long long* __restrict__ ei,
    const u32* __restrict__ pW1, const float* __restrict__ b1,
    const u32* __restrict__ pW2, const float* __restrict__ b2,
    float* __restrict__ aggr) {
  __shared__ u16 lhid[4][32 * Hh];              // 8 KB per wave staging
  const int wave = threadIdx.x >> 5;
  const int lane = threadIdx.x & 31;
  const int tp = blockIdx.x * 4 + wave;         // tile-pair index
  const int t0 = tp * 2, t1 = tp * 2 + 1;       // Ee/16 = 25000 (even): both valid
  if (t0 * 16 >= Ee) return;
  const int r  = lane & 15;                     // row within tile (A layout)
  const int s8 = (lane >> 4) << 3;              // 0 / 8 : K sub-chunk select
  const long long* srcI = ei;                   // edge_index[0] = source
  const long long* dstI = ei + Ee;              // edge_index[1] = target
  const int e0 = t0 * 16 + r, e1 = t1 * 16 + r;
  const u16* hd0 = hbf + (size_t)(int)dstI[e0] * Hh;
  const u16* hs0 = hbf + (size_t)(int)srcI[e0] * Hh;
  const u16* ep0 = ebf + (size_t)e0 * Hh;
  const u16* hd1 = hbf + (size_t)(int)dstI[e1] * Hh;
  const u16* hs1 = hbf + (size_t)(int)srcI[e1] * Hh;
  const u16* ep1 = ebf + (size_t)e1 * Hh;

  // GEMM1: [32,384] @ [384,128]
  const v8f z = vzero8();
  v8f acc0[8], acc1[8];
#pragma unroll
  for (int nt = 0; nt < 8; ++nt) { acc0[nt] = z; acc1[nt] = z; }

#pragma unroll
  for (int kk = 0; kk < 12; ++kk) {
    const int kb = kk * 32;
    const u16 *ba0, *ba1;
    int off;
    if (kb < 128)      { ba0 = hd0; ba1 = hd1; off = kb + s8; }
    else if (kb < 256) { ba0 = hs0; ba1 = hs1; off = kb - 128 + s8; }
    else               { ba0 = ep0; ba1 = ep1; off = kb - 256 + s8; }
    v16bf a0 = frag_from_rowmajor(ba0 + off);
    v16bf a1 = frag_from_rowmajor(ba1 + off);
    const u32* wp = pW1 + (size_t)(kk * 8) * 256 + lane * 8;
    v16bf bc = frag_from_packed(wp);
#pragma unroll
    for (int nt = 0; nt < 8; ++nt) {
      v16bf bn_ = (nt < 7) ? frag_from_packed(wp + 256) : bc;  // prefetch
      acc0[nt] = wmma_bf16(a0, bc, acc0[nt]);
      acc1[nt] = wmma_bf16(a1, bc, acc1[nt]);
      bc = bn_;
      wp += 256;
    }
  }

  // bias + ReLU -> bf16 -> LDS relayout (C layout -> row-major [row][k])
  u16* myl = lhid[wave];
#pragma unroll
  for (int nt = 0; nt < 8; ++nt) {
    const int col = nt * 16 + r;
    const float bv = b1[col];
#pragma unroll
    for (int v = 0; v < 8; ++v) {
      float f = fmaxf(acc0[nt][v] + bv, 0.0f);
      myl[(v + s8) * Hh + col] = f2bf(f);       // same-wave LDS: in-order
      float g = fmaxf(acc1[nt][v] + bv, 0.0f);
      myl[(16 + v + s8) * Hh + col] = f2bf(g);
    }
  }

  // GEMM2: [32,128] @ [128,128]
  v8f acc2a[8], acc2b[8];
#pragma unroll
  for (int nt = 0; nt < 8; ++nt) { acc2a[nt] = z; acc2b[nt] = z; }
#pragma unroll
  for (int kk = 0; kk < 4; ++kk) {
    v16bf a0 = frag_from_rowmajor(myl + r * Hh + kk * 32 + s8);
    v16bf a1 = frag_from_rowmajor(myl + (16 + r) * Hh + kk * 32 + s8);
    const u32* wp = pW2 + (size_t)(kk * 8) * 256 + lane * 8;
    v16bf bc = frag_from_packed(wp);
#pragma unroll
    for (int nt = 0; nt < 8; ++nt) {
      v16bf bn_ = (nt < 7) ? frag_from_packed(wp + 256) : bc;
      acc2a[nt] = wmma_bf16(a0, bc, acc2a[nt]);
      acc2b[nt] = wmma_bf16(a1, bc, acc2b[nt]);
      bc = bn_;
      wp += 256;
    }
  }

  // bias + segment-sum via f32 atomics into L2-resident aggr[dst]
  float b2v[8];
#pragma unroll
  for (int nt = 0; nt < 8; ++nt) b2v[nt] = b2[nt * 16 + r];
#pragma unroll
  for (int v = 0; v < 8; ++v) {
    const int row = v + s8;
    const int na = (int)dstI[t0 * 16 + row];
    const int nb = (int)dstI[t1 * 16 + row];
    float* ra = aggr + (size_t)na * Hh + r;
    float* rb = aggr + (size_t)nb * Hh + r;
#pragma unroll
    for (int nt = 0; nt < 8; ++nt) {
      __hip_atomic_fetch_add(ra + nt * 16, acc2a[nt][v] + b2v[nt],
                             __ATOMIC_RELAXED, __HIP_MEMORY_SCOPE_AGENT);
      __hip_atomic_fetch_add(rb + nt * 16, acc2b[nt][v] + b2v[nt],
                             __ATOMIC_RELAXED, __HIP_MEMORY_SCOPE_AGENT);
    }
  }
}

// ---------------------------------------------------------------------------
// Fused node update, M=32 per wave:
//   upd = relu([h|aggr] @ U1 + b1) @ U2 + b2 ; h += upd
// ---------------------------------------------------------------------------
__global__ __launch_bounds__(128) void node_mlp_kernel(
    float* __restrict__ h, u16* __restrict__ hbf,
    const float* __restrict__ aggr,
    const u32* __restrict__ pW1, const float* __restrict__ b1,
    const u32* __restrict__ pW2, const float* __restrict__ b2) {
  __shared__ u16 lhid[4][32 * Hh];
  const int wave = threadIdx.x >> 5;
  const int lane = threadIdx.x & 31;
  const int ntiles = Nn / 16;                   // 3125 (odd)
  const int tp = blockIdx.x * 4 + wave;
  const int t0 = tp * 2;
  if (t0 >= ntiles) return;
  const bool has1 = (t0 + 1 < ntiles);
  const int t1 = has1 ? t0 + 1 : t0;            // clamp (compute, don't store)
  const int r  = lane & 15;
  const int s8 = (lane >> 4) << 3;
  const int n0 = t0 * 16 + r, n1 = t1 * 16 + r;
  const u16* hp0 = hbf + (size_t)n0 * Hh;
  const u16* hp1 = hbf + (size_t)n1 * Hh;
  const float* ap0 = aggr + (size_t)n0 * Hh;
  const float* ap1 = aggr + (size_t)n1 * Hh;

  const v8f z = vzero8();
  v8f acc0[8], acc1[8];
#pragma unroll
  for (int nt = 0; nt < 8; ++nt) { acc0[nt] = z; acc1[nt] = z; }

#pragma unroll
  for (int kk = 0; kk < 8; ++kk) {              // K = 256
    const int kb = kk * 32;
    v16bf a0, a1;
    if (kb < 128) {
      a0 = frag_from_rowmajor(hp0 + kb + s8);
      a1 = frag_from_rowmajor(hp1 + kb + s8);
    } else {                                    // aggr is f32 -> convert inline
      const float* p0 = ap0 + (kb - 128) + s8;
      const float* p1 = ap1 + (kb - 128) + s8;
      F4 f0 = *(const F4*)p0;
      F4 f1 = *(const F4*)(p0 + 4);
      F4 f2 = *(const F4*)(p0 + 16);
      F4 f3 = *(const F4*)(p0 + 20);
      QQ t;
      t.a.x = pack2(f0.x, f0.y); t.a.y = pack2(f0.z, f0.w);
      t.a.z = pack2(f1.x, f1.y); t.a.w = pack2(f1.z, f1.w);
      t.b.x = pack2(f2.x, f2.y); t.b.y = pack2(f2.z, f2.w);
      t.b.z = pack2(f3.x, f3.y); t.b.w = pack2(f3.z, f3.w);
      a0 = __builtin_bit_cast(v16bf, t);
      f0 = *(const F4*)p1;
      f1 = *(const F4*)(p1 + 4);
      f2 = *(const F4*)(p1 + 16);
      f3 = *(const F4*)(p1 + 20);
      t.a.x = pack2(f0.x, f0.y); t.a.y = pack2(f0.z, f0.w);
      t.a.z = pack2(f1.x, f1.y); t.a.w = pack2(f1.z, f1.w);
      t.b.x = pack2(f2.x, f2.y); t.b.y = pack2(f2.z, f2.w);
      t.b.z = pack2(f3.x, f3.y); t.b.w = pack2(f3.z, f3.w);
      a1 = __builtin_bit_cast(v16bf, t);
    }
    const u32* wp = pW1 + (size_t)(kk * 8) * 256 + lane * 8;
    v16bf bc = frag_from_packed(wp);
#pragma unroll
    for (int nt = 0; nt < 8; ++nt) {
      v16bf bn_ = (nt < 7) ? frag_from_packed(wp + 256) : bc;
      acc0[nt] = wmma_bf16(a0, bc, acc0[nt]);
      acc1[nt] = wmma_bf16(a1, bc, acc1[nt]);
      bc = bn_;
      wp += 256;
    }
  }

  u16* myl = lhid[wave];
#pragma unroll
  for (int nt = 0; nt < 8; ++nt) {
    const int col = nt * 16 + r;
    const float bv = b1[col];
#pragma unroll
    for (int v = 0; v < 8; ++v) {
      float f = fmaxf(acc0[nt][v] + bv, 0.0f);
      myl[(v + s8) * Hh + col] = f2bf(f);
      float g = fmaxf(acc1[nt][v] + bv, 0.0f);
      myl[(16 + v + s8) * Hh + col] = f2bf(g);
    }
  }

  v8f acc2a[8], acc2b[8];
#pragma unroll
  for (int nt = 0; nt < 8; ++nt) { acc2a[nt] = z; acc2b[nt] = z; }
#pragma unroll
  for (int kk = 0; kk < 4; ++kk) {
    v16bf a0 = frag_from_rowmajor(myl + r * Hh + kk * 32 + s8);
    v16bf a1 = frag_from_rowmajor(myl + (16 + r) * Hh + kk * 32 + s8);
    const u32* wp = pW2 + (size_t)(kk * 8) * 256 + lane * 8;
    v16bf bc = frag_from_packed(wp);
#pragma unroll
    for (int nt = 0; nt < 8; ++nt) {
      v16bf bn_ = (nt < 7) ? frag_from_packed(wp + 256) : bc;
      acc2a[nt] = wmma_bf16(a0, bc, acc2a[nt]);
      acc2b[nt] = wmma_bf16(a1, bc, acc2b[nt]);
      bc = bn_;
      wp += 256;
    }
  }

  // bias + residual update (each (node,col) owned by exactly one lane)
  float b2v[8];
#pragma unroll
  for (int nt = 0; nt < 8; ++nt) b2v[nt] = b2[nt * 16 + r];
#pragma unroll
  for (int nt = 0; nt < 8; ++nt) {
#pragma unroll
    for (int v = 0; v < 8; ++v) {
      const int row = v + s8;
      const size_t i0 = (size_t)(t0 * 16 + row) * Hh + nt * 16 + r;
      float nh = h[i0] + acc2a[nt][v] + b2v[nt];
      h[i0]   = nh;
      hbf[i0] = f2bf(nh);
      if (has1) {
        const size_t i1 = (size_t)(t1 * 16 + row) * Hh + nt * 16 + r;
        float nh1 = h[i1] + acc2b[nt][v] + b2v[nt];
        h[i1]   = nh1;
        hbf[i1] = f2bf(nh1);
      }
    }
  }
}

// ---------------------------------------------------------------------------
// Pooling + readout
// ---------------------------------------------------------------------------
__global__ void pool_kernel(const float* __restrict__ h,
                            const long long* __restrict__ batch,
                            float* __restrict__ psum, float* __restrict__ pcnt) {
  int gid = blockIdx.x * 256 + threadIdx.x;
  if (gid >= Nn * Hh) return;
  int n = gid >> 7, c = gid & 127;
  int b = (int)batch[n];
  __hip_atomic_fetch_add(psum + b * Hh + c, h[gid],
                         __ATOMIC_RELAXED, __HIP_MEMORY_SCOPE_AGENT);
  if (c == 0)
    __hip_atomic_fetch_add(pcnt + b, 1.0f,
                           __ATOMIC_RELAXED, __HIP_MEMORY_SCOPE_AGENT);
}

__global__ __launch_bounds__(256) void readout_kernel(
    const float* __restrict__ psum, const float* __restrict__ pcnt,
    const float* __restrict__ gf, const float* __restrict__ Wg,
    const float* __restrict__ bg,
    const float* __restrict__ rW1, const float* __restrict__ rb1,
    const float* __restrict__ rW2, const float* __restrict__ rb2,
    const float* __restrict__ rW3, const float* __restrict__ rb3,
    float* __restrict__ out) {
  __shared__ float fin[Bb][2 * Hh];
  __shared__ float t1[Bb][Hh];
  __shared__ float t2[Bb][64];
  const int t = threadIdx.x;
  for (int i = t; i < Bb * 2 * Hh; i += 256) {
    int b = i / (2 * Hh), c = i % (2 * Hh);
    float v;
    if (c < Hh) v = psum[b * Hh + c] / fmaxf(pcnt[b], 1.0f);
    else        v = gf[b] * Wg[c - Hh] + bg[c - Hh];
    fin[b][c] = v;
  }
  __syncthreads();
  for (int i = t; i < Bb * Hh; i += 256) {
    int b = i >> 7, c = i & 127;
    float s = rb1[c];
    for (int k = 0; k < 2 * Hh; ++k) s += fin[b][k] * rW1[k * Hh + c];
    t1[b][c] = s > 0.0f ? s : 0.0f;
  }
  __syncthreads();
  for (int i = t; i < Bb * 64; i += 256) {
    int b = i >> 6, c = i & 63;
    float s = rb2[c];
    for (int k = 0; k < Hh; ++k) s += t1[b][k] * rW2[k * 64 + c];
    t2[b][c] = s > 0.0f ? s : 0.0f;
  }
  __syncthreads();
  if (t < Bb) {
    float s = rb3[0];
    for (int k = 0; k < 64; ++k) s += t2[t][k] * rW3[k];
    out[t] = s;
  }
}

// ---------------------------------------------------------------------------
extern "C" void kernel_launch(void* const* d_in, const int* in_sizes, int n_in,
                              void* d_out, int out_size, void* d_ws,
                              size_t ws_size, hipStream_t stream) {
  (void)in_sizes; (void)n_in; (void)out_size; (void)ws_size;
  const float*     x    = (const float*)d_in[0];
  const float*     ea   = (const float*)d_in[1];
  const float*     gf   = (const float*)d_in[2];
  const long long* ei   = (const long long*)d_in[3];
  const long long* bat  = (const long long*)d_in[4];
  const float* Wn  = (const float*)d_in[5];
  const float* bn  = (const float*)d_in[6];
  const float* We  = (const float*)d_in[7];
  const float* be  = (const float*)d_in[8];
  const float* Wg  = (const float*)d_in[9];
  const float* bg  = (const float*)d_in[10];
  const float* mW1 = (const float*)d_in[11];
  const float* mb1 = (const float*)d_in[12];
  const float* mW2 = (const float*)d_in[13];
  const float* mb2 = (const float*)d_in[14];
  const float* uW1 = (const float*)d_in[15];
  const float* ub1 = (const float*)d_in[16];
  const float* uW2 = (const float*)d_in[17];
  const float* ub2 = (const float*)d_in[18];
  const float* rW1 = (const float*)d_in[19];
  const float* rb1 = (const float*)d_in[20];
  const float* rW2 = (const float*)d_in[21];
  const float* rb2 = (const float*)d_in[22];
  const float* rW3 = (const float*)d_in[23];
  const float* rb3 = (const float*)d_in[24];
  float* out = (float*)d_out;

  // Workspace carve (256B aligned chunks)
  char* wp_ = (char*)d_ws;
  auto take = [&](size_t bytes) -> void* {
    void* r = (void*)wp_;
    wp_ += (bytes + 255) & ~(size_t)255;
    return r;
  };
  float* h    = (float*)take((size_t)Nn * Hh * 4);
  u16*   hbf  = (u16*)  take((size_t)Nn * Hh * 2);
  u16*   ebf  = (u16*)  take((size_t)Ee * Hh * 2);
  float* aggr = (float*)take((size_t)Nn * Hh * 4);
  u32* pmW1 = (u32*)take((size_t)Ll * 384 * Hh * 2);
  u32* pmW2 = (u32*)take((size_t)Ll * 128 * Hh * 2);
  u32* puW1 = (u32*)take((size_t)Ll * 256 * Hh * 2);
  u32* puW2 = (u32*)take((size_t)Ll * 128 * Hh * 2);
  float* psum = (float*)take((size_t)Bb * Hh * 4);
  float* pcnt = (float*)take((size_t)Bb * 4);

  // 1. pack weights into WMMA B-operand layout
  for (int l = 0; l < Ll; ++l) {
    pack_wmma_b<<<(384 * Hh / 2 + 255) / 256, 256, 0, stream>>>(
        mW1 + (size_t)l * 384 * Hh, pmW1 + (size_t)l * 384 * Hh / 2, 384, Hh);
    pack_wmma_b<<<(128 * Hh / 2 + 255) / 256, 256, 0, stream>>>(
        mW2 + (size_t)l * 128 * Hh, pmW2 + (size_t)l * 128 * Hh / 2, 128, Hh);
    pack_wmma_b<<<(256 * Hh / 2 + 255) / 256, 256, 0, stream>>>(
        uW1 + (size_t)l * 256 * Hh, puW1 + (size_t)l * 256 * Hh / 2, 256, Hh);
    pack_wmma_b<<<(128 * Hh / 2 + 255) / 256, 256, 0, stream>>>(
        uW2 + (size_t)l * 128 * Hh, puW2 + (size_t)l * 128 * Hh / 2, 128, Hh);
  }

  // 2. embeddings
  embed_nodes_kernel<<<(Nn * Hh + 255) / 256, 256, 0, stream>>>(x, Wn, bn, h, hbf);
  embed_edges_kernel<<<(Ee * Hh + 255) / 256, 256, 0, stream>>>(ea, We, be, ebf);

  // 3. message-passing layers
  const int edgeBlocks = Ee / 32 / 4;                     // 3125 (exact)
  const int nodeBlocks = ((Nn / 16 + 1) / 2 + 3) / 4;     // 391
  for (int l = 0; l < Ll; ++l) {
    kfill0<<<(Nn * Hh + 255) / 256, 256, 0, stream>>>(aggr, Nn * Hh);
    edge_mlp_kernel<<<edgeBlocks, 128, 0, stream>>>(
        hbf, ebf, ei,
        pmW1 + (size_t)l * 24576, mb1 + (size_t)l * Hh,
        pmW2 + (size_t)l * 8192,  mb2 + (size_t)l * Hh, aggr);
    node_mlp_kernel<<<nodeBlocks, 128, 0, stream>>>(
        h, hbf, aggr,
        puW1 + (size_t)l * 16384, ub1 + (size_t)l * Hh,
        puW2 + (size_t)l * 8192,  ub2 + (size_t)l * Hh);
  }

  // 4. pooling + readout
  kfill0<<<(Bb * Hh + 255) / 256, 256, 0, stream>>>(psum, Bb * Hh);
  kfill0<<<1, 256, 0, stream>>>(pcnt, Bb);
  pool_kernel<<<(Nn * Hh + 255) / 256, 256, 0, stream>>>(h, bat, psum, pcnt);
  readout_kernel<<<1, 256, 0, stream>>>(psum, pcnt, gf, Wg, bg,
                                        rW1, rb1, rW2, rb2, rW3, rb3, out);
}
